// GPT2Attention_11484742549737
// MI455X (gfx1250) — matrix operand order repaired and checked by reading, not compiled, not run on hardware
//
#include <hip/hip_runtime.h>
#include <hip/hip_bf16.h>
#include <stdint.h>

#define T_SEQ   2048
#define BATCH   4
#define NH      12
#define DK      64
#define DM      768
#define BT      (BATCH*T_SEQ)   // 8192
#define BHN     (BATCH*NH)      // 48

typedef __bf16 bf16_t;
typedef __attribute__((ext_vector_type(16))) __bf16        v16bf;
typedef __attribute__((ext_vector_type(8)))  __bf16        v8bf;
typedef __attribute__((ext_vector_type(8)))  float         v8f;
typedef __attribute__((ext_vector_type(8)))  unsigned int  v8u;

__device__ __forceinline__ bf16_t f2bf(float f) {
  unsigned u = __builtin_bit_cast(unsigned, f);
  unsigned r = u + 0x7FFFu + ((u >> 16) & 1u);   // round-to-nearest-even
  return __builtin_bit_cast(bf16_t, (unsigned short)(r >> 16));
}
__device__ __forceinline__ unsigned pack2(bf16_t a, bf16_t b) {
  return (unsigned)__builtin_bit_cast(unsigned short, a)
       | ((unsigned)__builtin_bit_cast(unsigned short, b) << 16);
}
__device__ __forceinline__ v16bf cat8(v8bf lo, v8bf hi) {
  v16bf r;
#pragma unroll
  for (int i = 0; i < 8; ++i) { r[i] = lo[i]; r[8+i] = hi[i]; }
  return r;
}
__device__ __forceinline__ v8f wmma_bf16(v16bf a, v16bf b, v8f c) {
  return __builtin_amdgcn_wmma_f32_16x16x32_bf16(false, a, false, b, (short)0, c,
                                                 false, false);
}

// -------- CDNA5 async global->LDS staging (ASYNCcnt-tracked DMA path) -----
// GVS form: mem_addr = SADDR64 + VGPR_u32_offset ; LDS dest = per-lane VGPR
__device__ __forceinline__ void async_load_b128(unsigned lds_off, unsigned g_off,
                                                const void* base) {
  asm volatile("global_load_async_to_lds_b128 %0, %1, %2 offset:0"
               :: "v"(lds_off), "v"(g_off), "s"(base) : "memory");
}
__device__ __forceinline__ void wait_async0() {
  asm volatile("s_wait_asynccnt 0x0" ::: "memory");
}
__device__ __forceinline__ unsigned lds_off32(const void* p) {
  // generic pointer to a __shared__ object: low 32 bits = LDS byte offset
  return (unsigned)(uintptr_t)p;
}

// ---------------- fp32 -> bf16 convert (4 elems/thread) -------------------
__global__ __launch_bounds__(256) void convert4_k(const float* __restrict__ in,
                                                  bf16_t* __restrict__ out, int n) {
  int i = (blockIdx.x * 256 + threadIdx.x) * 4;
  if (i < n) {
    float4 v = *(const float4*)(in + i);
    uint2 u;
    u.x = pack2(f2bf(v.x), f2bf(v.y));
    u.y = pack2(f2bf(v.z), f2bf(v.w));
    *(uint2*)(out + i) = u;
  }
}

// ------------- fp32 [R][C] -> bf16 [C][R] tiled transpose -----------------
__global__ __launch_bounds__(256) void transpose_convert_k(
    const float* __restrict__ in, bf16_t* __restrict__ out, int R, int C) {
  __shared__ float tile[32][33];
  const int tx = threadIdx.x & 31, ty = threadIdx.x >> 5;   // ty: 0..7
  const int cb = blockIdx.x * 32, rb = blockIdx.y * 32;
#pragma unroll
  for (int i = 0; i < 4; ++i) {
    int r = ty + i * 8;
    tile[r][tx] = in[(size_t)(rb + r) * C + cb + tx];
  }
  __syncthreads();
#pragma unroll
  for (int i = 0; i < 4; ++i) {
    int cl = ty + i * 8;
    out[(size_t)(cb + cl) * R + rb + tx] = f2bf(tile[tx][cl]);
  }
}

// ---------------- bf16 WMMA GEMM: C = A[M,K] * Bt[N,K]^T + bias -----------
// MODE 0: QKV projection, scatter to q[b][h][t][d], k[b][h][t][d], vt[b][h][d][t]
// MODE 1: output projection, fp32 out[M][N]
template<int MODE>
__global__ __launch_bounds__(256) void gemm_bf16_k(
    const bf16_t* __restrict__ A, const bf16_t* __restrict__ Bt,
    const float* __restrict__ bias,
    bf16_t* __restrict__ outQ, bf16_t* __restrict__ outK,
    bf16_t* __restrict__ outVt, float* __restrict__ outF, int N, int K) {
  __shared__ __align__(16) bf16_t As[128][40];
  __shared__ __align__(16) bf16_t Bs[128][40];
  const int tid  = threadIdx.x;
  const int wave = tid >> 5, lane = tid & 31;
  const int wm = wave >> 2, wn = wave & 3;     // 2x4 wave grid
  const int ln = lane & 15, lh = lane >> 4;
  const int mBlk = blockIdx.y * 128, nBlk = blockIdx.x * 128;

  const v8f zero8 = {0,0,0,0,0,0,0,0};
  v8f acc[4][2];
#pragma unroll
  for (int i = 0; i < 4; ++i) { acc[i][0] = zero8; acc[i][1] = zero8; }

  for (int k0 = 0; k0 < K; k0 += 32) {
    __syncthreads();
#pragma unroll
    for (int p = 0; p < 2; ++p) {
      int c = tid + p * 256;                    // 0..511
      int r = c >> 2, ch = (c & 3) * 8;
      async_load_b128(lds_off32(&As[r][ch]),
                      (unsigned)(((size_t)(mBlk + r) * K + k0 + ch) * 2), A);
      async_load_b128(lds_off32(&Bs[r][ch]),
                      (unsigned)(((size_t)(nBlk + r) * K + k0 + ch) * 2), Bt);
    }
    wait_async0();
    __syncthreads();

    v16bf bfrag[2];
#pragma unroll
    for (int nt = 0; nt < 2; ++nt) {
      int row = wn * 32 + nt * 16 + ln;
      bfrag[nt] = cat8(*(const v8bf*)(&Bs[row][16 * lh]),
                       *(const v8bf*)(&Bs[row][16 * lh + 8]));
    }
#pragma unroll
    for (int mt = 0; mt < 4; ++mt) {
      int row = wm * 64 + mt * 16 + ln;
      v16bf afrag = cat8(*(const v8bf*)(&As[row][16 * lh]),
                         *(const v8bf*)(&As[row][16 * lh + 8]));
#pragma unroll
      for (int nt = 0; nt < 2; ++nt)
        acc[mt][nt] = wmma_bf16(afrag, bfrag[nt], acc[mt][nt]);
    }
  }

#pragma unroll
  for (int mt = 0; mt < 4; ++mt)
#pragma unroll
    for (int nt = 0; nt < 2; ++nt) {
      int gn = nBlk + wn * 32 + nt * 16 + ln;
      float bv = bias[gn];
#pragma unroll
      for (int r = 0; r < 8; ++r) {
        int gm = mBlk + wm * 64 + mt * 16 + r + 8 * lh;
        float v = acc[mt][nt][r] + bv;
        if (MODE == 0) {
          int which = gn / DM;
          int rem = gn - which * DM;
          int h = rem >> 6, d = rem & 63;
          int bidx = gm >> 11, t = gm & (T_SEQ - 1);
          bf16_t bb = f2bf(v);
          size_t hoff = (size_t)bidx * NH + h;
          if (which == 0)      outQ [(hoff * T_SEQ + t) * DK + d] = bb;
          else if (which == 1) outK [(hoff * T_SEQ + t) * DK + d] = bb;
          else                 outVt[(hoff * DK + d) * T_SEQ + t] = bb;
        } else {
          outF[(size_t)gm * N + gn] = v;
        }
      }
    }
}

// ---------------- flash attention (computes S^T = K*Q^T) ------------------
// Q,K: [BH][T][64] bf16; Vt: [BH][64][T] bf16; Y: [B][T][768] bf16
__global__ __launch_bounds__(256) void attn_kernel(
    const bf16_t* __restrict__ Q, const bf16_t* __restrict__ Kd,
    const bf16_t* __restrict__ Vt, bf16_t* __restrict__ Y) {
  __shared__ __align__(16) bf16_t Qs[128][72];
  __shared__ __align__(16) bf16_t Ks[128][72];
  __shared__ __align__(16) bf16_t Vs[64][136];

  const int tid = threadIdx.x;
  const int w = tid >> 5, lane = tid & 31;
  const int ln = lane & 15, lh = lane >> 4;
  const int qb = blockIdx.x, bh = blockIdx.y;
  const int q0 = qb * 128;
  const bf16_t* Qh = Q  + (size_t)bh * T_SEQ * DK;
  const bf16_t* Kh = Kd + (size_t)bh * T_SEQ * DK;
  const bf16_t* Vh = Vt + (size_t)bh * DK * T_SEQ;
  const float L2E = 1.44269504f;

#pragma unroll
  for (int p = 0; p < 4; ++p) {                 // 128x64 Q tile (async DMA)
    int c = tid + p * 256;
    int r = c >> 3, ch = (c & 7) * 8;
    async_load_b128(lds_off32(&Qs[r][ch]),
                    (unsigned)(((size_t)(q0 + r) * DK + ch) * 2), Qh);
  }

  const v8f zero8 = {0,0,0,0,0,0,0,0};
  float mrow = -__builtin_inff(), lrow = 0.0f;
  v8f ot[4];
#pragma unroll
  for (int i = 0; i < 4; ++i) ot[i] = zero8;
  const int qg = q0 + w * 16 + ln;              // this lane's query

  for (int kb = 0; kb <= qb; ++kb) {
    const int k0 = kb * 128;
    __syncthreads();
#pragma unroll
    for (int p = 0; p < 4; ++p) {               // 128x64 K tile (async DMA)
      int c = tid + p * 256;
      int r = c >> 3, ch = (c & 7) * 8;
      async_load_b128(lds_off32(&Ks[r][ch]),
                      (unsigned)(((size_t)(k0 + r) * DK + ch) * 2), Kh);
    }
#pragma unroll
    for (int p = 0; p < 4; ++p) {               // 64x128 V^T tile (async DMA)
      int c = tid + p * 256;
      int r = c >> 4, ch = (c & 15) * 8;
      async_load_b128(lds_off32(&Vs[r][ch]),
                      (unsigned)(((size_t)r * T_SEQ + k0 + ch) * 2), Vh);
    }
    wait_async0();                              // also covers the Q tile (kb==0)
    __syncthreads();

    // S^T = K * Q^T : 8 key M-tiles x 1 query N-tile, two 32-d k-steps
    v8f st[8];
#pragma unroll
    for (int mt = 0; mt < 8; ++mt) st[mt] = zero8;
#pragma unroll
    for (int ks = 0; ks < 2; ++ks) {
      v16bf bq = cat8(*(const v8bf*)(&Qs[w*16 + ln][32*ks + 16*lh]),
                      *(const v8bf*)(&Qs[w*16 + ln][32*ks + 16*lh + 8]));
#pragma unroll
      for (int mt = 0; mt < 8; ++mt) {
        v16bf ak = cat8(*(const v8bf*)(&Ks[16*mt + ln][32*ks + 16*lh]),
                        *(const v8bf*)(&Ks[16*mt + ln][32*ks + 16*lh + 8]));
        st[mt] = wmma_bf16(ak, bq, st[mt]);
      }
    }

    // scale + causal mask + row max (keys are within-lane; one cross-half shfl)
    float rmax = -__builtin_inff();
#pragma unroll
    for (int mt = 0; mt < 8; ++mt)
#pragma unroll
      for (int r = 0; r < 8; ++r) {
        int kg = k0 + 16*mt + r + 8*lh;
        float s = st[mt][r] * 0.125f;           // 1/sqrt(64)
        s = (kg <= qg) ? s : -__builtin_inff();
        st[mt][r] = s;
        rmax = fmaxf(rmax, s);
      }
    rmax = fmaxf(rmax, __shfl_xor(rmax, 16, 32));
    float mnew  = fmaxf(mrow, rmax);
    float alpha = __builtin_exp2f((mrow - mnew) * L2E);

    float rsum = 0.0f;
    unsigned pp[8][4];                          // P^T as packed bf16 pairs
#pragma unroll
    for (int mt = 0; mt < 8; ++mt) {
      float pv[8];
#pragma unroll
      for (int r = 0; r < 8; ++r) {
        pv[r] = __builtin_exp2f((st[mt][r] - mnew) * L2E);
        rsum += pv[r];
      }
#pragma unroll
      for (int j = 0; j < 4; ++j)
        pp[mt][j] = pack2(f2bf(pv[2*j]), f2bf(pv[2*j + 1]));
    }
    rsum += __shfl_xor(rsum, 16, 32);
    lrow = lrow * alpha + rsum;
    mrow = mnew;
#pragma unroll
    for (int dt = 0; dt < 4; ++dt)
#pragma unroll
      for (int r = 0; r < 8; ++r) ot[dt][r] *= alpha;

    // O^T += V^T * P^T : B-fragments of P^T built in registers via shfl_xor(16)
#pragma unroll
    for (int ks2 = 0; ks2 < 4; ++ks2) {
      const int a = 2 * ks2, b2 = a + 1;
      unsigned pa[4], pb[4];
#pragma unroll
      for (int j = 0; j < 4; ++j) {
        pa[j] = (unsigned)__shfl_xor((int)pp[a][j],  16, 32);
        pb[j] = (unsigned)__shfl_xor((int)pp[b2][j], 16, 32);
      }
      v8u t;
#pragma unroll
      for (int j = 0; j < 4; ++j) {
        t[j]     = lh ? pb[j]     : pp[a][j];   // keys +0..7 (from h=0 source)
        t[4 + j] = lh ? pp[b2][j] : pa[j];      // keys +8..15 (from h=1 source)
      }
      v16bf pfrag = __builtin_bit_cast(v16bf, t);
#pragma unroll
      for (int dt = 0; dt < 4; ++dt) {
        v16bf av = cat8(*(const v8bf*)(&Vs[16*dt + ln][32*ks2 + 16*lh]),
                        *(const v8bf*)(&Vs[16*dt + ln][32*ks2 + 16*lh + 8]));
        ot[dt] = wmma_bf16(av, pfrag, ot[dt]);
      }
    }
  }

  const float inv = 1.0f / lrow;
  const int bidx = bh / NH, h = bh % NH;
  bf16_t* yrow = Y + ((size_t)bidx * T_SEQ + qg) * DM + h * DK;
#pragma unroll
  for (int dt = 0; dt < 4; ++dt)
#pragma unroll
    for (int r = 0; r < 8; ++r) {
      int d = 16 * dt + r + 8 * lh;
      yrow[d] = f2bf(ot[dt][r] * inv);
    }
}

// --------------------------------------------------------------------------
extern "C" void kernel_launch(void* const* d_in, const int* in_sizes, int n_in,
                              void* d_out, int out_size, void* d_ws, size_t ws_size,
                              hipStream_t stream) {
  (void)in_sizes; (void)n_in; (void)out_size; (void)ws_size;
  const float* x   = (const float*)d_in[0];
  // d_in[1] = mask (unused by the reference module)
  const float* caw = (const float*)d_in[2];
  const float* cab = (const float*)d_in[3];
  const float* wow = (const float*)d_in[4];
  const float* wob = (const float*)d_in[5];
  float* out = (float*)d_out;

  char* ws = (char*)d_ws;
  size_t off = 0;
  bf16_t* xb  = (bf16_t*)(ws + off); off += (size_t)BT * DM * 2;      // x bf16, reused as Y
  bf16_t* wqT = (bf16_t*)(ws + off); off += (size_t)3 * DM * DM * 2;  // c_attn_w^T
  bf16_t* woT = (bf16_t*)(ws + off); off += (size_t)DM * DM * 2;      // wo_w^T
  bf16_t* q   = (bf16_t*)(ws + off); off += (size_t)BHN * T_SEQ * DK * 2;
  bf16_t* k   = (bf16_t*)(ws + off); off += (size_t)BHN * T_SEQ * DK * 2;
  bf16_t* vt  = (bf16_t*)(ws + off); off += (size_t)BHN * DK * T_SEQ * 2;

  convert4_k<<<(BT * DM) / 1024, 256, 0, stream>>>(x, xb, BT * DM);
  transpose_convert_k<<<dim3((3 * DM) / 32, DM / 32), 256, 0, stream>>>(caw, wqT, DM, 3 * DM);
  transpose_convert_k<<<dim3(DM / 32, DM / 32), 256, 0, stream>>>(wow, woT, DM, DM);

  gemm_bf16_k<0><<<dim3((3 * DM) / 128, BT / 128), 256, 0, stream>>>(
      xb, wqT, cab, q, k, vt, nullptr, 3 * DM, DM);

  attn_kernel<<<dim3(T_SEQ / 128, BHN), 256, 0, stream>>>(q, k, vt, xb /* Y */);

  gemm_bf16_k<1><<<dim3(DM / 128, BT / 128), 256, 0, stream>>>(
      xb, woT, wob, nullptr, nullptr, nullptr, out, DM, DM);
}
